// GatedLinearAttention_23965917512228
// MI455X (gfx1250) — compile-verified, matrix operand order checked by reading, not run on hardware
//
#include <hip/hip_runtime.h>

// ---------------------------------------------------------------------------
// Problem constants (from reference)
// ---------------------------------------------------------------------------
// B=2, T=2048, HID=2048, H=16, DK=64, DV=128, C=16, LOWR=16
// M = B*T = 4096 rows.  Fused projection GEMM: [4096,2048] x [2048,6144]
//   cols 0..1023   : q  (H*DK)
//   cols 1024..2047: k  (H*DK)
//   cols 2048..4095: v  (H*DV)   (overwritten with o by the GLA kernel)
//   cols 4096..6143: g  (H*DV)
// Output GEMM: [4096,2048] x [2048,2048] -> d_out (fp32)

#define MROWS   4096
#define KDIM    2048
#define NCAT    6144
#define NOUT    2048
#define GLA_SCALE 0.125f   // DK^-0.5

typedef __attribute__((ext_vector_type(16))) __bf16 v16bf;
typedef __attribute__((ext_vector_type(8)))  float  v8f;

__device__ __forceinline__ unsigned short f2bf(float f) {
  unsigned int u = __float_as_uint(f);
  u += 0x7fffu + ((u >> 16) & 1u);
  return (unsigned short)(u >> 16);
}

// gfx1250 async global->LDS DMA: 16B per lane, no data VGPRs, ASYNCcnt-tracked.
// GV mode: VDST = LDS byte offset (low 32 bits of flat shared address per the
// LDS aperture rule addr[31:0]), VADDR = 64-bit global address, no SADDR.
__device__ __forceinline__ void async_copy_b128(const void* g, void* lds) {
  unsigned lds_off = (unsigned)(unsigned long long)lds;
  asm volatile("global_load_async_to_lds_b128 %0, %1, off"
               :: "v"(lds_off), "v"(g) : "memory");
}
__device__ __forceinline__ void wait_async0() {
  asm volatile("s_wait_asynccnt 0x0" ::: "memory");
}

// ---------------------------------------------------------------------------
// fp32 -> bf16 elementwise convert (x activations)
// ---------------------------------------------------------------------------
__global__ __launch_bounds__(256) void cvt_f32_bf16_kernel(
    const float* __restrict__ in, unsigned short* __restrict__ out, long n) {
  long i = ((long)blockIdx.x * 256 + threadIdx.x) * 4;
  if (i + 3 < n) {
    float4 v = *(const float4*)(in + i);
    out[i + 0] = f2bf(v.x);
    out[i + 1] = f2bf(v.y);
    out[i + 2] = f2bf(v.z);
    out[i + 3] = f2bf(v.w);
  }
}

// ---------------------------------------------------------------------------
// Transpose [K,Nw] fp32 -> [Nw,K] bf16 (weights, so WMMA B operand is N-major)
// ---------------------------------------------------------------------------
__global__ __launch_bounds__(256) void transpose_cvt_kernel(
    const float* __restrict__ W, unsigned short* __restrict__ out,
    int K, int Nw) {
  __shared__ float tile[32][33];
  int kb = blockIdx.y * 32, nb = blockIdx.x * 32;
  int tx = threadIdx.x, ty = threadIdx.y;   // block (32,8)
#pragma unroll
  for (int i = 0; i < 32; i += 8)
    tile[ty + i][tx] = W[(long)(kb + ty + i) * Nw + (nb + tx)];
  __syncthreads();
#pragma unroll
  for (int i = 0; i < 32; i += 8)
    out[(long)(nb + ty + i) * K + (kb + tx)] = f2bf(tile[tx][ty + i]);
}

// ---------------------------------------------------------------------------
// bf16 WMMA GEMM:  C[M,N](f32) = A[M,K](bf16,row-major) * Bt[N,K]^T
// Block tile 128x128, K-step 64, 256 threads = 8 waves (4x2 wave grid),
// each wave: 2x4 fragments of 16x16. Double-buffered LDS, pitch 72
// (144B rows -> conflict-free b128 fragment reads). Tile staging uses
// global_load_async_to_lds_b128 (no data VGPRs -> no spills).
// ---------------------------------------------------------------------------
#define GT_M 128
#define GT_K 64
#define LDT  72

__device__ __forceinline__ v16bf load_frag(const unsigned short* p) {
  union { v16bf v; uint4 u[2]; } f;
  f.u[0] = *(const uint4*)(p);        // halves 0..7  : K = khalf*8 + 0..7
  f.u[1] = *(const uint4*)(p + 16);   // halves 8..15 : K = khalf*8 + 16..23
  return f.v;
}

__global__ __launch_bounds__(256) void gemm_bf16_wmma_kernel(
    const unsigned short* __restrict__ A,   // [M,K] bf16
    const unsigned short* __restrict__ Bt,  // [N,K] bf16 (B transposed)
    float* __restrict__ C, int K, int ldc) {
  __shared__ __align__(16) unsigned short As[2][GT_M * LDT];
  __shared__ __align__(16) unsigned short Bs[2][GT_M * LDT];

  const int tid  = threadIdx.x;
  const int lane = tid & 31;
  const int w    = tid >> 5;
  const int wm   = w >> 1;   // 0..3 : 32-row band
  const int wn   = w & 1;    // 0..1 : 64-col band

  const long m0 = (long)blockIdx.y * GT_M;
  const long n0 = (long)blockIdx.x * GT_M;
  const int  c0 = tid * 4;            // this thread's 4 of 1024 16B chunks
  const int  nk = K / GT_K;

  const unsigned short* Agb = A  + m0 * K;
  const unsigned short* Bgb = Bt + n0 * K;

  // preload tile 0 (async DMA straight into LDS)
#pragma unroll
  for (int i = 0; i < 4; ++i) {
    int ch = c0 + i, r = ch >> 3, kc = (ch & 7) * 8;
    async_copy_b128(&Agb[(long)r * K + kc], &As[0][r * LDT + kc]);
    async_copy_b128(&Bgb[(long)r * K + kc], &Bs[0][r * LDT + kc]);
  }
  wait_async0();
  __syncthreads();

  v8f acc[2][4];
#pragma unroll
  for (int fm = 0; fm < 2; ++fm)
#pragma unroll
    for (int fn = 0; fn < 4; ++fn)
      acc[fm][fn] = (v8f){0.f, 0.f, 0.f, 0.f, 0.f, 0.f, 0.f, 0.f};

  for (int kt = 0; kt < nk; ++kt) {
    const int cur = kt & 1, nxt = cur ^ 1;
    // kick off async DMA of the next tile into the other buffer
    if (kt + 1 < nk) {
      const unsigned short* Ag = Agb + (kt + 1) * GT_K;
      const unsigned short* Bg = Bgb + (kt + 1) * GT_K;
#pragma unroll
      for (int i = 0; i < 4; ++i) {
        int ch = c0 + i, r = ch >> 3, kc = (ch & 7) * 8;
        async_copy_b128(&Ag[(long)r * K + kc], &As[nxt][r * LDT + kc]);
        async_copy_b128(&Bg[(long)r * K + kc], &Bs[nxt][r * LDT + kc]);
      }
    }
    if (kt + 2 < nk) {  // gfx1250 global_prefetch_b8 for tile kt+2 into L2
      int r = c0 >> 3, kc = (c0 & 7) * 8;
      __builtin_prefetch(&Agb[(kt + 2) * GT_K + (long)r * K + kc], 0, 1);
      __builtin_prefetch(&Bgb[(kt + 2) * GT_K + (long)r * K + kc], 0, 1);
    }

    // compute on current buffer: two k=32 sub-steps
#pragma unroll
    for (int s = 0; s < 2; ++s) {
      const int kb = s * 32 + (lane >> 4) * 8;   // khalf selects K sub-lane set
      v16bf afr[2], bfr[4];
#pragma unroll
      for (int fm = 0; fm < 2; ++fm) {
        int row = wm * 32 + fm * 16 + (lane & 15);
        afr[fm] = load_frag(&As[cur][row * LDT + kb]);
      }
#pragma unroll
      for (int fn = 0; fn < 4; ++fn) {
        int row = wn * 64 + fn * 16 + (lane & 15);
        bfr[fn] = load_frag(&Bs[cur][row * LDT + kb]);
      }
#pragma unroll
      for (int fm = 0; fm < 2; ++fm)
#pragma unroll
        for (int fn = 0; fn < 4; ++fn)
          acc[fm][fn] = __builtin_amdgcn_wmma_f32_16x16x32_bf16(
              false, afr[fm], false, bfr[fn], (short)0, acc[fm][fn],
              false, false);
    }

    // next tile must be fully in LDS before anyone reads it next iteration
    wait_async0();
    __syncthreads();
  }

  // store: VGPR r -> row (r + 8*khalf), lane&15 -> col  (16x16 f32 C layout)
  const int cn    = lane & 15;
  const int rbase = (lane >> 4) * 8;
#pragma unroll
  for (int fm = 0; fm < 2; ++fm)
#pragma unroll
    for (int fn = 0; fn < 4; ++fn) {
      long row0 = m0 + wm * 32 + fm * 16 + rbase;
      long col  = n0 + wn * 64 + fn * 16 + cn;
#pragma unroll
      for (int r = 0; r < 8; ++r)
        C[(row0 + r) * ldc + col] = acc[fm][fn][r];
    }
}

// ---------------------------------------------------------------------------
// gk = log_sigmoid(x @ Wgk1 @ Wgk2 + bgk2) / 16      (low-rank, fp32)
// one block per row m; wave w reduces ranks 2w, 2w+1 with wave32 shuffles
// ---------------------------------------------------------------------------
__global__ __launch_bounds__(256) void gk_kernel(
    const float* __restrict__ x, const float* __restrict__ Wgk1,
    const float* __restrict__ Wgk2, const float* __restrict__ bgk2,
    float* __restrict__ gkout) {
  __shared__ float xrow[KDIM];
  __shared__ float low[16];
  const int m = blockIdx.x;
  const float* xr = x + (long)m * KDIM;
  for (int k = threadIdx.x; k < KDIM; k += 256) xrow[k] = xr[k];
  __syncthreads();

  const int w = threadIdx.x >> 5, lane = threadIdx.x & 31;
  float s0 = 0.f, s1 = 0.f;
  for (int k = lane; k < KDIM; k += 32) {
    float xv = xrow[k];
    s0 += xv * Wgk1[k * 16 + 2 * w];
    s1 += xv * Wgk1[k * 16 + 2 * w + 1];
  }
#pragma unroll
  for (int off = 16; off > 0; off >>= 1) {
    s0 += __shfl_down(s0, off, 32);
    s1 += __shfl_down(s1, off, 32);
  }
  if (lane == 0) { low[2 * w] = s0; low[2 * w + 1] = s1; }
  __syncthreads();

  float l[16];
#pragma unroll
  for (int r = 0; r < 16; ++r) l[r] = low[r];
  for (int j = threadIdx.x; j < 1024; j += 256) {
    float acc = bgk2[j];
#pragma unroll
    for (int r = 0; r < 16; ++r) acc += l[r] * Wgk2[r * 1024 + j];
    float ls = fminf(acc, 0.f) - log1pf(__expf(-fabsf(acc)));
    gkout[(long)m * 1024 + j] = ls * (1.0f / 16.0f);
  }
}

// ---------------------------------------------------------------------------
// Chunked GLA scan: one block per (b,h), 128 threads (thread = DV column e),
// S[64][128] carried in LDS across 128 chunks of 16. o overwrites v in qkvg.
// ---------------------------------------------------------------------------
#define CPAD 68
__global__ __launch_bounds__(128) void gla_kernel(
    float* __restrict__ qkvg, const float* __restrict__ gk) {
  __shared__ float S[64 * 128];
  __shared__ float gc[16 * CPAD];
  __shared__ float qe[16 * CPAD];
  __shared__ float ke[16 * CPAD];
  __shared__ float kd[16 * CPAD];
  __shared__ float Am[16 * 16];
  __shared__ float decay[64];

  const int bh = blockIdx.x;        // 0..31
  const int b = bh >> 4, h = bh & 15;
  const int tid = threadIdx.x;

  for (int i = tid; i < 64 * 128; i += 128) S[i] = 0.f;
  __syncthreads();

  const long rowbase = (long)b * 2048;
  float v[16], o[16];

  for (int n = 0; n < 128; ++n) {
    const long m0 = rowbase + n * 16;
    if (tid < 64) {                  // per-column-d prep
      const int d = tid;
      float c = 0.f;
#pragma unroll
      for (int cc = 0; cc < 16; ++cc) {
        c += gk[(m0 + cc) * 1024 + h * 64 + d];
        gc[cc * CPAD + d] = c;
      }
      decay[d] = __expf(c);
      const float glast = c;
#pragma unroll
      for (int cc = 0; cc < 16; ++cc) {
        float g = gc[cc * CPAD + d];
        float q = qkvg[(m0 + cc) * 6144 + h * 64 + d];
        float k = qkvg[(m0 + cc) * 6144 + 1024 + h * 64 + d];
        qe[cc * CPAD + d] = q * __expf(g) * GLA_SCALE;
        ke[cc * CPAD + d] = k * __expf(-g);
        kd[cc * CPAD + d] = k * __expf(glast - g);
      }
    }
    __syncthreads();

    // A = tril(qe ke^T), 256 entries over 128 threads
#pragma unroll
    for (int t = tid; t < 256; t += 128) {
      int cc = t >> 4, mm = t & 15;
      float a = 0.f;
      if (mm <= cc) {
#pragma unroll
        for (int d = 0; d < 64; ++d) a += qe[cc * CPAD + d] * ke[mm * CPAD + d];
      }
      Am[cc * 16 + mm] = a;
    }
    __syncthreads();

    {
      const int e = tid;
#pragma unroll
      for (int cc = 0; cc < 16; ++cc)
        v[cc] = qkvg[(m0 + cc) * 6144 + 2048 + h * 128 + e];
      // o_intra
#pragma unroll
      for (int cc = 0; cc < 16; ++cc) {
        float a = 0.f;
        for (int mm = 0; mm <= cc; ++mm) a += Am[cc * 16 + mm] * v[mm];
        o[cc] = a;
      }
      // o_inter (uses old S) fused with state update
      for (int d = 0; d < 64; ++d) {
        float s = S[d * 128 + e];
        float kv = 0.f;
#pragma unroll
        for (int cc = 0; cc < 16; ++cc) {
          o[cc] += qe[cc * CPAD + d] * s;
          kv += kd[cc * CPAD + d] * v[cc];
        }
        S[d * 128 + e] = decay[d] * s + kv;
      }
#pragma unroll
      for (int cc = 0; cc < 16; ++cc)
        qkvg[(m0 + cc) * 6144 + 2048 + h * 128 + e] = o[cc];
    }
    __syncthreads();
  }
}

// ---------------------------------------------------------------------------
// RMSNorm over DV + SiLU(g) gate, output bf16 for the final GEMM
// one block per (m,h), 128 threads (thread = e)
// ---------------------------------------------------------------------------
__global__ __launch_bounds__(128) void norm_gate_kernel(
    const float* __restrict__ qkvg, const float* __restrict__ norm_w,
    unsigned short* __restrict__ og) {
  __shared__ float red[128];
  const int mh = blockIdx.x;
  const int m = mh >> 4, h = mh & 15;
  const int e = threadIdx.x;
  float o = qkvg[(long)m * 6144 + 2048 + h * 128 + e];
  red[e] = o * o;
  __syncthreads();
#pragma unroll
  for (int s = 64; s > 0; s >>= 1) {
    if (e < s) red[e] += red[e + s];
    __syncthreads();
  }
  float r = rsqrtf(red[0] * (1.f / 128.f) + 1e-5f);
  float g = qkvg[(long)m * 6144 + 4096 + h * 128 + e];
  float val = o * r * norm_w[e] * (g / (1.f + __expf(-g)));
  og[(long)m * 2048 + h * 128 + e] = f2bf(val);
}

// ---------------------------------------------------------------------------
// Host launcher
// ---------------------------------------------------------------------------
extern "C" void kernel_launch(void* const* d_in, const int* in_sizes, int n_in,
                              void* d_out, int out_size, void* d_ws,
                              size_t ws_size, hipStream_t stream) {
  const float* x      = (const float*)d_in[0];
  const float* Wq     = (const float*)d_in[1];
  const float* Wk     = (const float*)d_in[2];
  const float* Wv     = (const float*)d_in[3];
  const float* Wg     = (const float*)d_in[4];
  const float* Wgk1   = (const float*)d_in[5];
  const float* Wgk2   = (const float*)d_in[6];
  const float* bgk2   = (const float*)d_in[7];
  const float* Wo     = (const float*)d_in[8];
  const float* norm_w = (const float*)d_in[9];
  float* out = (float*)d_out;

  // workspace layout (bytes)
  char* ws = (char*)d_ws;
  unsigned short* xb    = (unsigned short*)(ws);                       // 16 MiB
  unsigned short* wcat  = (unsigned short*)(ws + (size_t)16777216);    // 24 MiB
  unsigned short* wot   = (unsigned short*)(ws + (size_t)41943040);    //  8 MiB
  float*          qkvg  = (float*)         (ws + (size_t)50331648);    // 96 MiB
  float*          gkbuf = (float*)         (ws + (size_t)150994944);   // 16 MiB
  unsigned short* og    = (unsigned short*)(ws + (size_t)167772160);   // 16 MiB

  // 1) x -> bf16
  cvt_f32_bf16_kernel<<<MROWS * KDIM / 1024, 256, 0, stream>>>(
      x, xb, (long)MROWS * KDIM);

  // 2) weight transpose+convert into concatenated [6144,2048] and [2048,2048]
  dim3 tb(32, 8);
  transpose_cvt_kernel<<<dim3(1024 / 32, KDIM / 32), tb, 0, stream>>>(
      Wq, wcat, KDIM, 1024);
  transpose_cvt_kernel<<<dim3(1024 / 32, KDIM / 32), tb, 0, stream>>>(
      Wk, wcat + (size_t)1024 * KDIM, KDIM, 1024);
  transpose_cvt_kernel<<<dim3(2048 / 32, KDIM / 32), tb, 0, stream>>>(
      Wv, wcat + (size_t)2048 * KDIM, KDIM, 2048);
  transpose_cvt_kernel<<<dim3(2048 / 32, KDIM / 32), tb, 0, stream>>>(
      Wg, wcat + (size_t)4096 * KDIM, KDIM, 2048);
  transpose_cvt_kernel<<<dim3(2048 / 32, KDIM / 32), tb, 0, stream>>>(
      Wo, wot, KDIM, 2048);

  // 3) fused projection GEMM (WMMA bf16 -> f32): qkvg[4096,6144]
  gemm_bf16_wmma_kernel<<<dim3(NCAT / 128, MROWS / 128), 256, 0, stream>>>(
      xb, wcat, qkvg, KDIM, NCAT);

  // 4) low-rank gate path gk[4096,1024]
  gk_kernel<<<MROWS, 256, 0, stream>>>(x, Wgk1, Wgk2, bgk2, gkbuf);

  // 5) chunked GLA scan (writes o over the v slice of qkvg)
  gla_kernel<<<32, 128, 0, stream>>>(qkvg, gkbuf);

  // 6) RMSNorm + SiLU gate -> bf16 [4096,2048]
  norm_gate_kernel<<<MROWS * 16, 128, 0, stream>>>(qkvg, norm_w, og);

  // 7) output GEMM (WMMA) -> d_out fp32 [4096,2048]
  gemm_bf16_wmma_kernel<<<dim3(NOUT / 128, MROWS / 128), 256, 0, stream>>>(
      og, wot, out, KDIM, NOUT);
}
